// SingleStreamBlock_17179869654
// MI455X (gfx1250) — matrix operand order repaired and checked by reading, not compiled, not run on hardware
//
#include <hip/hip_runtime.h>
#include <hip/hip_bf16.h>

// ---------------- constants ----------------
static constexpr int SEQ  = 2048;
static constexpr int DIM  = 3072;
static constexpr int NH   = 24;
static constexpr int DHD  = 128;
static constexpr int QKVW = 9216;    // 3*DIM
static constexpr int N1W  = 21504;   // QKVW + 4*DIM   (w1 output width)
static constexpr int N2W  = 15360;   // DIM + 4*DIM    (concat width / w2 input)

// GEMM tiling
static constexpr int BM = 256;       // rows per block
static constexpr int BN = 128;       // cols per block
static constexpr int LDSP = 17;      // padded row stride in dwords (16 data + 1 pad)

// ---------------- bf16 helpers ----------------
typedef __attribute__((ext_vector_type(16))) __bf16 v16bf;
typedef __attribute__((ext_vector_type(8)))  float  v8f;

union Frag { v16bf v; unsigned u[8]; };

__device__ __forceinline__ unsigned short f2b(float f) {
  unsigned u = __builtin_bit_cast(unsigned, f);
  unsigned r = (u + 0x7FFFu + ((u >> 16) & 1u)) >> 16;   // round-nearest-even
  return (unsigned short)r;
}
__device__ __forceinline__ float b2f(unsigned short h) {
  unsigned u = ((unsigned)h) << 16;
  return __builtin_bit_cast(float, u);
}
__device__ __forceinline__ v8f wmma_bf16(const Frag& a, const Frag& b, v8f c) {
  return __builtin_amdgcn_wmma_f32_16x16x32_bf16(false, a.v, false, b.v,
                                                 (short)0, c, false, false);
}

// ---------------- Tensor Data Mover (gfx1250) ----------------
#if __has_builtin(__builtin_amdgcn_tensor_load_to_lds) && __has_builtin(__builtin_amdgcn_s_wait_tensorcnt)
#define USE_TDM 1
#else
#define USE_TDM 0
#endif

// Async-DMA a BM x 32 bf16 tile (row stride K elements) from global into LDS,
// with TDM-inserted +1-dword padding after every 64B row (matches LDSP=17).
__device__ __forceinline__ void tdm_load_A(const unsigned short* gsrc,
                                           unsigned lds_addr, int K) {
#if USE_TDM
  typedef __attribute__((ext_vector_type(4))) unsigned v4u;
  typedef __attribute__((ext_vector_type(8))) int v8i;
  typedef __attribute__((ext_vector_type(4))) int v4i;
  unsigned long long ga = (unsigned long long)(uintptr_t)gsrc;
  // D# group 0: count=1 | lds_addr | global_addr[56:0] | type=2
  v4u g0 = { 1u, lds_addr, (unsigned)ga,
             (unsigned)((ga >> 32) & 0x1FFFFFFu) | 0x80000000u };
  // D# group 1: data_size=2B, pad_enable, pad_interval=3 (16 dwords), pad_amount=0 (1 dword)
  //   tensor_dim0=K, tensor_dim1=SEQ, tile_dim0=32, tile_dim1=BM, dim0_stride=K
  v8i g1 = { (int)((1u << 16) | (1u << 20) | (3u << 22)),
             (int)(((unsigned)K & 0xFFFFu) << 16),
             (int)(((unsigned)K >> 16) | ((unsigned)SEQ << 16)),
             (int)(32u << 16),
             (int)BM,
             (int)K,
             0, 0 };
  v4i z4 = {0, 0, 0, 0};
#if __clang_major__ >= 23
  v8i z8 = {0, 0, 0, 0, 0, 0, 0, 0};
  __builtin_amdgcn_tensor_load_to_lds(g0, g1, z4, z4, z8, 0);
#else
  __builtin_amdgcn_tensor_load_to_lds(g0, g1, z4, z4, 0);
#endif
#else
  (void)gsrc; (void)lds_addr; (void)K;
#endif
}
__device__ __forceinline__ void wait_tensorcnt0() {
#if USE_TDM
  __builtin_amdgcn_s_wait_tensorcnt(0);
#endif
}

// ---------------- K0a: silu(vec) ----------------
__global__ void silu_kernel(const float* __restrict__ vec, float* __restrict__ sv) {
  int i = blockIdx.x * 256 + threadIdx.x;
  float v = vec[i];
  sv[i] = v / (1.f + __expf(-v));
}

// ---------------- K0b: m = silu(vec) @ w_mod + b_mod ----------------
__global__ void gemv_kernel(const float* __restrict__ sv, const float* __restrict__ wmod,
                            const float* __restrict__ bmod, float* __restrict__ mv) {
  int n = blockIdx.x * 256 + threadIdx.x;   // n < 3*DIM
  float acc = bmod[n];
  for (int k = 0; k < DIM; ++k) acc += sv[k] * wmod[(size_t)k * (3 * DIM) + n];
  mv[n] = acc;
}

// ---------------- K1: LayerNorm + modulate -> bf16 x_mod ----------------
__global__ __launch_bounds__(256)
void ln_mod_kernel(const float* __restrict__ x, const float* __restrict__ mv,
                   unsigned short* __restrict__ xmod) {
  const int row = blockIdx.x, tid = threadIdx.x;
  const float* xr = x + (size_t)row * DIM;
  float s = 0.f, s2 = 0.f;
  for (int c = tid; c < DIM; c += 256) { float v = xr[c]; s += v; s2 += v * v; }
  __shared__ float rs[8], rs2[8];
  for (int off = 16; off >= 1; off >>= 1) {
    s  += __shfl_xor(s, off, 32);
    s2 += __shfl_xor(s2, off, 32);
  }
  if ((tid & 31) == 0) { rs[tid >> 5] = s; rs2[tid >> 5] = s2; }
  __syncthreads();
  float S = 0.f, S2 = 0.f;
  for (int i = 0; i < 8; ++i) { S += rs[i]; S2 += rs2[i]; }
  float mu  = S * (1.f / DIM);
  float var = S2 * (1.f / DIM) - mu * mu;
  float inv = rsqrtf(var + 1e-6f);
  for (int c = tid; c < DIM; c += 256) {
    float xn = (xr[c] - mu) * inv;
    float y  = (1.f + mv[DIM + c]) * xn + mv[c];   // scale, shift
    xmod[(size_t)row * DIM + c] = f2b(y);
  }
}

// ---------------- GEMM: C = A(bf16, MxK) @ B(f32->bf16, KxN) ----------------
// 512 threads = 16 waves; block tile BM(256) x BN(128); double-buffered LDS.
// A tile staged via TDM (async DMA) when available; B tile converted f32->bf16
// through registers (prefetched one K-step ahead).
// MODE 0: epilogue -> qkv bf16 buffer + gelu(mlp) into concat buffer
// MODE 1: epilogue -> out = x + gate * (acc + b2)   (f32)
template <int MODE>
__global__ __launch_bounds__(512)
void gemm_bf16_kernel(const unsigned short* __restrict__ A,
                      const float* __restrict__ B, int N, int K,
                      const float* __restrict__ bias,
                      unsigned short* __restrict__ o_qkv,
                      unsigned short* __restrict__ o_a2,
                      float* __restrict__ o_f32,
                      const float* __restrict__ xres,
                      const float* __restrict__ gate) {
  __shared__ unsigned As4[2][BM][LDSP];   // [buf][row][k/2] packed bf16 pairs
  __shared__ unsigned Bs4[2][BN][LDSP];   // [buf][col][k/2] packed bf16 pairs
  const int tid = threadIdx.x;
  const int wv = tid >> 5, lane = tid & 31;
  const int hl = lane >> 4, lr = lane & 15;
  const int brow = blockIdx.y * BM;
  const int bcol = blockIdx.x * BN;

  // per-thread B-tile task: 2 consecutive k-rows x 4 cols
  const int br = tid >> 5;          // 0..15 -> k rows 2*br, 2*br+1
  const int bc = (tid & 31) * 4;    // col within tile

  auto loadB = [&](int k0, float4& f0, float4& f1) {
    f0 = *(const float4*)(B + (size_t)(k0 + 2 * br)     * N + bcol + bc);
    f1 = *(const float4*)(B + (size_t)(k0 + 2 * br + 1) * N + bcol + bc);
  };
  auto storeB = [&](int buf, const float4& f0, const float4& f1) {
    Bs4[buf][bc + 0][br] = (unsigned)f2b(f0.x) | ((unsigned)f2b(f1.x) << 16);
    Bs4[buf][bc + 1][br] = (unsigned)f2b(f0.y) | ((unsigned)f2b(f1.y) << 16);
    Bs4[buf][bc + 2][br] = (unsigned)f2b(f0.z) | ((unsigned)f2b(f1.z) << 16);
    Bs4[buf][bc + 3][br] = (unsigned)f2b(f0.w) | ((unsigned)f2b(f1.w) << 16);
  };
#if !USE_TDM
  auto loadA = [&](int k0, uint4& a0, uint4& a1) {
    int u0 = tid * 2, u1 = tid * 2 + 1;
    a0 = *(const uint4*)(A + (size_t)(brow + (u0 >> 2)) * K + k0 + (u0 & 3) * 8);
    a1 = *(const uint4*)(A + (size_t)(brow + (u1 >> 2)) * K + k0 + (u1 & 3) * 8);
  };
  auto storeA = [&](int buf, const uint4& a0, const uint4& a1) {
    int u0 = tid * 2, u1 = tid * 2 + 1;
    *(uint4*)(&As4[buf][u0 >> 2][(u0 & 3) * 4]) = a0;
    *(uint4*)(&As4[buf][u1 >> 2][(u1 & 3) * 4]) = a1;
  };
#endif

  v8f acc[8] = {};

  // ---- preload tile 0 ----
#if USE_TDM
  if (wv == 0)
    tdm_load_A(A + (size_t)brow * K, (unsigned)(uintptr_t)&As4[0][0][0], K);
#else
  { uint4 a0, a1; loadA(0, a0, a1); storeA(0, a0, a1); }
#endif
  { float4 f0, f1; loadB(0, f0, f1); storeB(0, f0, f1); }
#if USE_TDM
  if (wv == 0) wait_tensorcnt0();
#endif
  __syncthreads();

  const int nsteps = K / 32;
  for (int s = 0; s < nsteps; ++s) {
    const int cur = s & 1, nxt = cur ^ 1;
    const int k0 = s * 32;
    const bool more = (s + 1 < nsteps);
    float4 nf0, nf1;
#if !USE_TDM
    uint4 na0, na1;
#endif
    if (more) {
      loadB(k0 + 32, nf0, nf1);
#if USE_TDM
      if (wv == 0)
        tdm_load_A(A + (size_t)brow * K + (k0 + 32),
                   (unsigned)(uintptr_t)&As4[nxt][0][0], K);
#else
      loadA(k0 + 32, na0, na1);
#endif
    }
    // ---- compute from LDS[cur] ----
    Frag af;
    {
      int ar = wv * 16 + lr;
      #pragma unroll
      for (int j = 0; j < 4; ++j) {
        af.u[j]     = As4[cur][ar][hl * 8 + j];
        af.u[4 + j] = As4[cur][ar][hl * 8 + 4 + j];
      }
    }
    #pragma unroll
    for (int nt = 0; nt < 8; ++nt) {
      Frag bfr;
      int col = nt * 16 + lr;
      #pragma unroll
      for (int j = 0; j < 8; ++j) bfr.u[j] = Bs4[cur][col][hl * 8 + j];
      acc[nt] = wmma_bf16(af, bfr, acc[nt]);
    }
    // ---- stage next tile ----
    if (more) {
      storeB(nxt, nf0, nf1);
#if USE_TDM
      if (wv == 0) wait_tensorcnt0();
#else
      storeA(nxt, na0, na1);
#endif
    }
    __syncthreads();
  }

  // epilogue: C layout — lane = n + 16*(m>>3), vgpr v = m&7
  const int r0 = brow + wv * 16 + hl * 8;
  #pragma unroll
  for (int nt = 0; nt < 8; ++nt) {
    int col = bcol + nt * 16 + lr;
    float bv = bias[col];
    #pragma unroll
    for (int v = 0; v < 8; ++v) {
      int row = r0 + v;
      float val = acc[nt][v] + bv;
      if (MODE == 0) {
        if (col < QKVW) {
          o_qkv[(size_t)row * QKVW + col] = f2b(val);
        } else {
          float t = val;
          float g = 0.5f * t * (1.f + tanhf(0.7978845608028654f * (t + 0.044715f * t * t * t)));
          o_a2[(size_t)row * N2W + (col - QKVW + DIM)] = f2b(g);
        }
      } else {
        o_f32[(size_t)row * DIM + col] = xres[(size_t)row * DIM + col] + gate[col] * val;
      }
    }
  }
}

// ---------------- K3: RMS-norm + scale + RoPE; V transposed ----------------
__global__ __launch_bounds__(128)
void rmsnorm_rope_kernel(const unsigned short* __restrict__ qkv,
                         const float* __restrict__ pe,
                         const float* __restrict__ qs, const float* __restrict__ ks,
                         unsigned short* __restrict__ qb, unsigned short* __restrict__ kb,
                         unsigned short* __restrict__ vt) {
  const int row = blockIdx.x, h = blockIdx.y, d = threadIdx.x;
  const unsigned short* base = qkv + (size_t)row * QKVW + h * DHD;
  float q = b2f(base[d]);
  float k = b2f(base[DIM + d]);
  float v = b2f(base[2 * DIM + d]);
  __shared__ float red[2][4];
  float sq = q * q, sk = k * k;
  #pragma unroll
  for (int off = 16; off >= 1; off >>= 1) {
    sq += __shfl_xor(sq, off, 32);
    sk += __shfl_xor(sk, off, 32);
  }
  if ((d & 31) == 0) { red[0][d >> 5] = sq; red[1][d >> 5] = sk; }
  __syncthreads();
  float tq = red[0][0] + red[0][1] + red[0][2] + red[0][3];
  float tk = red[1][0] + red[1][1] + red[1][2] + red[1][3];
  q *= rsqrtf(tq * (1.f / DHD) + 1e-6f) * qs[d];
  k *= rsqrtf(tk * (1.f / DHD) + 1e-6f) * ks[d];
  __shared__ float qsh[DHD], ksh[DHD];
  qsh[d] = q; ksh[d] = k;
  __syncthreads();
  const int i = d >> 1, a = d & 1;
  const float* per = pe + ((size_t)row * (DHD / 2) + i) * 4 + a * 2;
  float qo = per[0] * qsh[2 * i] + per[1] * qsh[2 * i + 1];
  float ko = per[0] * ksh[2 * i] + per[1] * ksh[2 * i + 1];
  qb[((size_t)h * SEQ + row) * DHD + d] = f2b(qo * 0.08838834764831845f); // Dh^-1/2 folded in
  kb[((size_t)h * SEQ + row) * DHD + d] = f2b(ko);
  vt[((size_t)h * DHD + d) * SEQ + row] = f2b(v);
}

// ---------------- K4: flash attention (1 wave = 16 queries) ----------------
__global__ __launch_bounds__(128)
void attn_kernel(const unsigned short* __restrict__ qb,
                 const unsigned short* __restrict__ kb,
                 const unsigned short* __restrict__ vt,
                 unsigned short* __restrict__ a2) {
  __shared__ __align__(16) unsigned short pst[4][16][32];   // per-wave P staging
  const int wv = threadIdx.x >> 5, lane = threadIdx.x & 31;
  const int hl = lane >> 4, lr = lane & 15;
  const int h = blockIdx.y;
  const int q0 = (blockIdx.x * 4 + wv) * 16;
  const unsigned short* Q  = qb + ((size_t)h * SEQ + q0) * DHD;
  const unsigned short* Kh = kb + (size_t)h * SEQ * DHD;
  const unsigned short* Vh = vt + (size_t)h * DHD * SEQ;

  // Q fragments: 4 chunks of K=32 over Dh
  Frag qf[4];
  #pragma unroll
  for (int c = 0; c < 4; ++c) {
    const unsigned short* qr = Q + (size_t)lr * DHD + c * 32;
    #pragma unroll
    for (int j = 0; j < 4; ++j) {
      qf[c].u[j]     = *(const unsigned*)(qr + 16 * hl + 2 * j);
      qf[c].u[4 + j] = *(const unsigned*)(qr + 16 * hl + 8 + 2 * j);
    }
  }
  v8f o[8] = {};
  float mrow[8], lrow[8];
  #pragma unroll
  for (int v = 0; v < 8; ++v) { mrow[v] = -1e30f; lrow[v] = 0.f; }

  for (int k0 = 0; k0 < SEQ; k0 += 32) {
    // S = Q @ K^T for 32 keys (two 16x16 tiles)
    v8f s0 = {}, s1 = {};
    #pragma unroll
    for (int c = 0; c < 4; ++c) {
      Frag b0, b1;
      const unsigned short* kr = Kh + (size_t)(k0 + lr) * DHD + c * 32 + 16 * hl;
      #pragma unroll
      for (int j = 0; j < 8; ++j) {
        b0.u[j] = *(const unsigned*)(kr + 2 * j);
        b1.u[j] = *(const unsigned*)(kr + 16 * DHD + 2 * j);
      }
      s0 = wmma_bf16(qf[c], b0, s0);
      s1 = wmma_bf16(qf[c], b1, s1);
    }
    // online softmax (rows live in 16-lane groups)
    float scl[8];
    #pragma unroll
    for (int v = 0; v < 8; ++v) {
      float mx = fmaxf(s0[v], s1[v]);
      #pragma unroll
      for (int off = 8; off >= 1; off >>= 1) mx = fmaxf(mx, __shfl_xor(mx, off, 16));
      float mn = fmaxf(mrow[v], mx);
      scl[v] = __expf(mrow[v] - mn);
      mrow[v] = mn;
      float p0 = __expf(s0[v] - mn);
      float p1 = __expf(s1[v] - mn);
      s0[v] = p0; s1[v] = p1;
      float rsum = p0 + p1;
      #pragma unroll
      for (int off = 8; off >= 1; off >>= 1) rsum += __shfl_xor(rsum, off, 16);
      lrow[v] = lrow[v] * scl[v] + rsum;
    }
    // stage P (C layout) -> LDS -> reload in A-fragment layout
    #pragma unroll
    for (int v = 0; v < 8; ++v) {
      int m = v + 8 * hl;
      pst[wv][m][lr]      = f2b(s0[v]);
      pst[wv][m][16 + lr] = f2b(s1[v]);
    }
    asm volatile("" ::: "memory");   // keep DS order; LDS is in-order per wave
    Frag pf;
    #pragma unroll
    for (int j = 0; j < 4; ++j) {
      pf.u[j]     = *(const unsigned*)(&pst[wv][lr][16 * hl + 2 * j]);
      pf.u[4 + j] = *(const unsigned*)(&pst[wv][lr][16 * hl + 8 + 2 * j]);
    }
    // O = O*scale + P @ V
    #pragma unroll
    for (int nt = 0; nt < 8; ++nt) {
      #pragma unroll
      for (int v = 0; v < 8; ++v) o[nt][v] *= scl[v];
      Frag vf;
      const unsigned short* vr = Vh + (size_t)(nt * 16 + lr) * SEQ + k0 + 16 * hl;
      #pragma unroll
      for (int j = 0; j < 8; ++j) vf.u[j] = *(const unsigned*)(vr + 2 * j);
      o[nt] = wmma_bf16(pf, vf, o[nt]);
    }
  }
  // write attn part of concat buffer
  #pragma unroll
  for (int nt = 0; nt < 8; ++nt) {
    #pragma unroll
    for (int v = 0; v < 8; ++v) {
      int row = q0 + v + 8 * hl;
      a2[(size_t)row * N2W + h * DHD + nt * 16 + lr] = f2b(o[nt][v] / lrow[v]);
    }
  }
}

// ---------------- launch ----------------
extern "C" void kernel_launch(void* const* d_in, const int* in_sizes, int n_in,
                              void* d_out, int out_size, void* d_ws, size_t ws_size,
                              hipStream_t stream) {
  (void)in_sizes; (void)n_in; (void)out_size; (void)ws_size;
  const float* x       = (const float*)d_in[0];
  const float* vec     = (const float*)d_in[1];
  const float* pe      = (const float*)d_in[2];
  const float* w_mod   = (const float*)d_in[3];
  const float* b_mod   = (const float*)d_in[4];
  const float* w1      = (const float*)d_in[5];
  const float* b1      = (const float*)d_in[6];
  const float* w2      = (const float*)d_in[7];
  const float* b2      = (const float*)d_in[8];
  const float* q_scale = (const float*)d_in[9];
  const float* k_scale = (const float*)d_in[10];
  float* out = (float*)d_out;

  char* w = (char*)d_ws;
  size_t off = 0;
  auto alloc = [&](size_t bytes) {
    void* p = w + off;
    off = (off + bytes + 255) & ~(size_t)255;
    return p;
  };
  float*          sv   = (float*)alloc((size_t)DIM * 4);
  float*          mv   = (float*)alloc((size_t)3 * DIM * 4);
  unsigned short* xmod = (unsigned short*)alloc((size_t)SEQ * DIM * 2);
  unsigned short* qkvb = (unsigned short*)alloc((size_t)SEQ * QKVW * 2);
  unsigned short* qbuf = (unsigned short*)alloc((size_t)NH * SEQ * DHD * 2);
  unsigned short* kbuf = (unsigned short*)alloc((size_t)NH * SEQ * DHD * 2);
  unsigned short* vtb  = (unsigned short*)alloc((size_t)NH * SEQ * DHD * 2);
  unsigned short* a2   = (unsigned short*)alloc((size_t)SEQ * N2W * 2);

  silu_kernel<<<DIM / 256, 256, 0, stream>>>(vec, sv);
  gemv_kernel<<<(3 * DIM) / 256, 256, 0, stream>>>(sv, w_mod, b_mod, mv);
  ln_mod_kernel<<<SEQ, 256, 0, stream>>>(x, mv, xmod);
  gemm_bf16_kernel<0><<<dim3(N1W / BN, SEQ / BM), 512, 0, stream>>>(
      xmod, w1, N1W, DIM, b1, qkvb, a2, nullptr, nullptr, nullptr);
  rmsnorm_rope_kernel<<<dim3(SEQ, NH), DHD, 0, stream>>>(
      qkvb, pe, q_scale, k_scale, qbuf, kbuf, vtb);
  attn_kernel<<<dim3(SEQ / 64, NH), 128, 0, stream>>>(qbuf, kbuf, vtb, a2);
  gemm_bf16_kernel<1><<<dim3(DIM / BN, SEQ / BM), 512, 0, stream>>>(
      a2, w2, DIM, N2W, b2, nullptr, nullptr, out, x, mv + 2 * DIM);
}
